// RNNClassifier_2087354106156
// MI455X (gfx1250) — matrix operand order repaired and checked by reading, not compiled, not run on hardware
//
#include <hip/hip_runtime.h>

#define VOCAB  50257
#define EMBED  256
#define HIDDEN 512
#define OUTC   10
#define BATCH  64
#define SEQL   512

typedef __attribute__((ext_vector_type(16))) __bf16 v16bf;
typedef __attribute__((ext_vector_type(8)))  __bf16 v8bf;
typedef __attribute__((ext_vector_type(8)))  float  v8f;
typedef __attribute__((ext_vector_type(4)))  float  v4f;

static __device__ __forceinline__ v16bf cat8(v8bf lo, v8bf hi) {
  return __builtin_shufflevector(lo, hi, 0,1,2,3,4,5,6,7,8,9,10,11,12,13,14,15);
}

// Async 16-byte global -> LDS copy (GLOBAL_LOAD_ASYNC_TO_LDS_B128, ASYNCcnt).
static __device__ __forceinline__ void async_cp16(void* ldsp, const void* gp) {
  asm volatile("global_load_async_to_lds_b128 %0, %1, off"
               :: "v"((unsigned)(unsigned long long)ldsp), "v"(gp)
               : "memory");
}

static __device__ __forceinline__ void async_wait0() {
#if __has_builtin(__builtin_amdgcn_s_wait_asynccnt)
  __builtin_amdgcn_s_wait_asynccnt(0);
#else
  asm volatile("s_wait_asynccnt 0x0" ::: "memory");
#endif
}

// ---------------- Kernel A: weight conversion + bias fuse ----------------
__global__ void rnn_prep_kernel(const float* __restrict__ Wih,
                                const float* __restrict__ Whh,
                                const float* __restrict__ b_ih,
                                const float* __restrict__ b_hh,
                                __bf16* __restrict__ Wihb,
                                __bf16* __restrict__ Whhb,
                                float* __restrict__ bias) {
  int i = blockIdx.x * blockDim.x + threadIdx.x;
  if (i < HIDDEN * EMBED)  Wihb[i] = (__bf16)Wih[i];
  if (i < HIDDEN * HIDDEN) Whhb[i] = (__bf16)Whh[i];
  if (i < HIDDEN)          bias[i] = b_ih[i] + b_hh[i];
}

// ---------------- Kernel B: embedding gather + input projection GEMM ----------------
// xw[s][b][h] = sum_e emb[x[b][s]][e] * Wih[h][e] + (b_ih+b_hh)[h]
// M = B*S = 32768 (rows m = b*SEQ+s), N = 512, K = 256. One 16x16 tile per wave.
__global__ __launch_bounds__(256) void rnn_proj_kernel(const int* __restrict__ x,
                                                       const float* __restrict__ emb,
                                                       const __bf16* __restrict__ Wihb,
                                                       const float* __restrict__ bias,
                                                       float* __restrict__ xw) {
  const int lane = threadIdx.x & 31;
  const int wave = threadIdx.x >> 5;
  const int tile = blockIdx.x * 8 + wave;     // 65536 tiles total
  const int m0 = (tile >> 5) * 16;            // 2048 M-tiles
  const int n0 = (tile & 31) * 16;            // 32 N-tiles
  const int row = lane & 15;
  const int hi  = lane >> 4;                  // lane half selects K sub-range

  const int tok = x[m0 + row];                // token for this A row
  const float*  arow = emb  + (long)tok * EMBED;
  const __bf16* brow = Wihb + (long)(n0 + row) * EMBED;

  v8f acc = {};
  #pragma unroll
  for (int kk = 0; kk < EMBED; kk += 32) {
    // A operand: lane<16 holds K {kk..kk+7, kk+16..kk+23}; lane>=16 shifted by 8
    const float* ap = arow + kk + hi * 8;
    v4f f0 = *(const v4f*)(ap +  0);
    v4f f1 = *(const v4f*)(ap +  4);
    v4f f2 = *(const v4f*)(ap + 16);
    v4f f3 = *(const v4f*)(ap + 20);
    v16bf a;
    #pragma unroll
    for (int j = 0; j < 4; ++j) {
      a[j]      = (__bf16)f0[j];
      a[4 + j]  = (__bf16)f1[j];
      a[8 + j]  = (__bf16)f2[j];
      a[12 + j] = (__bf16)f3[j];
    }
    // B operand: lane%16 = column, lane<16 holds K {kk..kk+7, kk+16..kk+23}; lane>=16 shifted by 8
    v16bf b = cat8(*(const v8bf*)(brow + kk + hi * 8), *(const v8bf*)(brow + kk + hi * 8 + 16));
    acc = __builtin_amdgcn_wmma_f32_16x16x32_bf16(false, a, false, b,
                                                  (short)0, acc, false, false);
  }

  const float bn = bias[n0 + row];
  #pragma unroll
  for (int r = 0; r < 8; ++r) {
    int m = m0 + r + hi * 8;                  // global row = b*SEQ + s
    int b = m >> 9;                           // / SEQ
    int s = m & (SEQL - 1);
    xw[((long)(s * BATCH + b)) * HIDDEN + n0 + row] = acc[r] + bn;
  }
}

// ---------------- Kernel C: recurrence + head ----------------
// 4 blocks, each owns 16 batch rows; 8 waves each own a 64-wide N-slice.
// h in LDS as bf16 (padded rows -> conflict-free ds_load_b128 A-operand reads).
// xw_t slices (32 KB/step) double-buffered in dynamic LDS, filled by async
// global->LDS DMA one step ahead of the WMMA pipeline.
__global__ __launch_bounds__(256) void rnn_scan_kernel(const float* __restrict__ xw,
                                                       const __bf16* __restrict__ Whhb,
                                                       const float* __restrict__ Wd,
                                                       const float* __restrict__ bd,
                                                       float* __restrict__ out) {
  __shared__ __bf16 hb[16 * 520];             // h (bf16), padded rows
  extern __shared__ float xwls[];             // 2 x (16*512) f32 staging buffers

  const int lane = threadIdx.x & 31;
  const int wave = threadIdx.x >> 5;
  const int b0   = blockIdx.x * 16;           // batch tile base
  const int row  = lane & 15;
  const int hi   = lane >> 4;
  const int nw0  = wave * 64;                 // wave's N base (4 tiles of 16)

  // Prefetch xw[t=0] into buffer 0 (coalesced 16B async chunks; 2048 chunks).
  {
    const float* g0 = xw + (long)b0 * HIDDEN; // (0*BATCH + b0)*HIDDEN
    #pragma unroll
    for (int j = 0; j < 8; ++j) {
      int c = threadIdx.x + j * 256;          // chunk of 4 floats
      async_cp16(xwls + c * 4, g0 + c * 4);
    }
  }
  for (int i = threadIdx.x; i < 16 * 520; i += 256) hb[i] = (__bf16)0.0f;
  async_wait0();
  __syncthreads();

  // Hoisted, t-invariant Whh row pointers (one per N-tile of this wave).
  const __bf16* bbase[4];
  #pragma unroll
  for (int nt = 0; nt < 4; ++nt)
    bbase[nt] = Whhb + (long)(nw0 + nt * 16 + row) * HIDDEN + hi * 16;

  for (int t = 0; t < SEQL; ++t) {
    float* lscur = xwls + (t & 1) * (16 * 512);

    // Kick off async DMA of next timestep's xw slice into the other buffer.
    if (t + 1 < SEQL) {
      float* lsnxt = xwls + ((t + 1) & 1) * (16 * 512);
      const float* g = xw + (long)((t + 1) * BATCH + b0) * HIDDEN;
      #pragma unroll
      for (int j = 0; j < 8; ++j) {
        int c = threadIdx.x + j * 256;
        async_cp16(lsnxt + c * 4, g + c * 4);
      }
    }

    // C operand from LDS staging (fuses the "+ xw_t" add).
    v8f acc[4];
    #pragma unroll
    for (int nt = 0; nt < 4; ++nt) {
      int n = nw0 + nt * 16 + row;
      #pragma unroll
      for (int r = 0; r < 8; ++r)
        acc[nt][r] = lscur[(r + hi * 8) * 512 + n];
    }

    // h @ Whh^T : A = h (bf16, LDS), B = Whh row-major bf16 (L2-resident)
    #pragma unroll 2
    for (int kk = 0; kk < HIDDEN; kk += 32) {
      const __bf16* ap = &hb[row * 520 + kk + hi * 16];
      v8bf a0 = *(const v8bf*)(ap);
      v8bf a1 = *(const v8bf*)(ap + 8);
      v16bf a = cat8(a0, a1);
      #pragma unroll
      for (int nt = 0; nt < 4; ++nt) {
        v16bf b = *(const v16bf*)(bbase[nt] + kk);
        acc[nt] = __builtin_amdgcn_wmma_f32_16x16x32_bf16(false, a, false, b,
                                                          (short)0, acc[nt], false, false);
      }
    }
    __syncthreads();                          // all reads of h done

    #pragma unroll
    for (int nt = 0; nt < 4; ++nt) {
      int n = nw0 + nt * 16 + row;
      #pragma unroll
      for (int r = 0; r < 8; ++r) {
        float hv = tanhf(acc[nt][r]);
        int m = r + hi * 8;
        hb[m * 520 + n] = (__bf16)hv;
        // Final h (f32) parked in staging buffer 0 (dead at t==SEQL-1, which is odd).
        if (t == SEQL - 1) xwls[m * 512 + n] = hv;
      }
    }
    async_wait0();                            // my DMA chunks for t+1 landed
    __syncthreads();                          // h + staged xw visible to all
  }

  // Head: out[b0+m][o] = hT[m] . Wd[o] + bd[o]   (160 dot products)
  if (threadIdx.x < 160) {
    int m = threadIdx.x / OUTC;
    int o = threadIdx.x % OUTC;
    float s = bd[o];
    const float* wd = Wd + (long)o * HIDDEN;
    for (int k = 0; k < HIDDEN; ++k) s += xwls[m * 512 + k] * wd[k];
    out[(b0 + m) * OUTC + o] = s;
  }
}

// ---------------- Host launch ----------------
extern "C" void kernel_launch(void* const* d_in, const int* in_sizes, int n_in,
                              void* d_out, int out_size, void* d_ws, size_t ws_size,
                              hipStream_t stream) {
  const int*   x    = (const int*)  d_in[0];
  const float* emb  = (const float*)d_in[1];
  const float* Wih  = (const float*)d_in[2];
  const float* Whh  = (const float*)d_in[3];
  const float* b_ih = (const float*)d_in[4];
  const float* b_hh = (const float*)d_in[5];
  const float* Wd   = (const float*)d_in[6];
  const float* bd   = (const float*)d_in[7];
  float* out = (float*)d_out;

  // workspace layout
  char* ws = (char*)d_ws;
  const size_t xw_bytes   = (size_t)SEQL * BATCH * HIDDEN * sizeof(float);   // 64 MB
  const size_t whhb_bytes = (size_t)HIDDEN * HIDDEN * sizeof(short);         // 512 KB
  const size_t wihb_bytes = (size_t)HIDDEN * EMBED * sizeof(short);          // 256 KB
  float*  xw    = (float*)(ws);
  __bf16* Whhb  = (__bf16*)(ws + xw_bytes);
  __bf16* Wihb  = (__bf16*)(ws + xw_bytes + whhb_bytes);
  float*  biasf = (float*)(ws + xw_bytes + whhb_bytes + wihb_bytes);

  // A: convert weights to bf16, fuse biases (covers 512*512 elements)
  rnn_prep_kernel<<<(HIDDEN * HIDDEN + 255) / 256, 256, 0, stream>>>(
      Wih, Whh, b_ih, b_hh, Wihb, Whhb, biasf);

  // B: gather + input projection. 65536 tiles / 8 waves per block
  rnn_proj_kernel<<<8192, 256, 0, stream>>>(x, emb, Wihb, biasf, xw);

  // C: recurrence + head. One block per 16 batch rows; 64 KB dynamic LDS staging.
  rnn_scan_kernel<<<BATCH / 16, 256, 2 * 16 * 512 * sizeof(float), stream>>>(
      xw, Whhb, Wd, bd, out);
}
